// SwinTransformerBlock_34462817583441
// MI455X (gfx1250) — compile-verified, matrix operand order checked by reading, not run on hardware
//
#include <hip/hip_runtime.h>
#include <math.h>

// ---------------------------------------------------------------------------
// SwinTransformerBlock (hierarchical/sparse variant) for gfx1250 (MI455X).
// All 7 GEMMs run on v_wmma_f32_16x16x32_f16 (f16 inputs, f32 accumulate).
// Attention (head_dim==1) is VALU exp/log work with LDS-staged k/v and
// wave32 shuffle reductions. Two-pass attention around the global
// entropy -> km_k scalar.  Deterministic: no atomics, fixed reduction trees.
// ---------------------------------------------------------------------------

typedef __attribute__((ext_vector_type(16))) _Float16 v16h;
typedef __attribute__((ext_vector_type(8)))  _Float16 v8h;
typedef __attribute__((ext_vector_type(8)))  float    v8f;

#define M_TOK   8192      // 2 * 64 * 64 tokens
#define CDIM    128
#define NWIN    128       // 2 * 8 * 8 windows
#define NTOK    64        // tokens per window
#define NHEAD   128
#define QKV_LD  384
#define SCALE_A 0.08838834764831845f   // 128^-0.5
#define GATE_LOGIT (-2.1972245773362196f) // log(0.1/0.9): sigmoid(x)>0.1
#define LN_EPS_F 1e-5f

// ------------------------------- wave utils --------------------------------
__device__ __forceinline__ float wred_sum(float v) {
#pragma unroll
  for (int o = 16; o; o >>= 1) v += __shfl_xor(v, o, 32);
  return v;
}
__device__ __forceinline__ float wred_max(float v) {
#pragma unroll
  for (int o = 16; o; o >>= 1) v = fmaxf(v, __shfl_xor(v, o, 32));
  return v;
}
__device__ __forceinline__ float wred_min(float v) {
#pragma unroll
  for (int o = 16; o; o >>= 1) v = fminf(v, __shfl_xor(v, o, 32));
  return v;
}
__device__ __forceinline__ unsigned sortable_u(float f) {
  unsigned u = __float_as_uint(f);
  return (u & 0x80000000u) ? ~u : (u | 0x80000000u);
}
__device__ __forceinline__ unsigned long long sx64(unsigned long long v, int m) {
  int lo = __shfl_xor((int)(unsigned)(v & 0xffffffffull), m, 32);
  int hi = __shfl_xor((int)(unsigned)(v >> 32), m, 32);
  return ((unsigned long long)(unsigned)hi << 32) | (unsigned)lo;
}

// --------------------------- weight f32 -> f16 -----------------------------
__global__ void k_f2h(const float* __restrict__ in, _Float16* __restrict__ out, int n) {
  int i = blockIdx.x * 256 + threadIdx.x;
  if (i < n) out[i] = (_Float16)in[i];
}

// ------------------- LayerNorm1 + window partition -------------------------
// wave per token; 128 channels = 4 per lane.
__global__ __launch_bounds__(256) void k_ln1_part(
    const float* __restrict__ x, const float* __restrict__ w,
    const float* __restrict__ b, float* __restrict__ xw,
    _Float16* __restrict__ xwh) {
  int lane = threadIdx.x & 31, wave = threadIdx.x >> 5;
  int t = blockIdx.x * 8 + wave;
  int bidx = t >> 12, l = t & 4095;
  int hh = l >> 6, wc = l & 63;
  int win = bidx * 64 + (hh >> 3) * 8 + (wc >> 3);
  int n = (hh & 7) * 8 + (wc & 7);
  const float* xr = x + (size_t)t * CDIM;
  float v[4]; float s = 0.f;
#pragma unroll
  for (int i = 0; i < 4; ++i) { v[i] = xr[lane + 32 * i]; s += v[i]; }
  s = wred_sum(s);
  float mu = s * (1.f / CDIM);
  float q = 0.f;
#pragma unroll
  for (int i = 0; i < 4; ++i) { float d = v[i] - mu; q += d * d; }
  q = wred_sum(q);
  float rstd = rsqrtf(q * (1.f / CDIM) + LN_EPS_F);
  size_t ob = ((size_t)win * NTOK + n) * CDIM;
#pragma unroll
  for (int i = 0; i < 4; ++i) {
    int c = lane + 32 * i;
    float o = (v[i] - mu) * rstd * w[c] + b[c];
    xw[ob + c] = o;
    xwh[ob + c] = (_Float16)o;
  }
}

// ------------------------- hierarchical gate -------------------------------
__global__ __launch_bounds__(128) void k_win_imp(
    const float* __restrict__ xw, const float* __restrict__ hg_w,
    const float* __restrict__ hg_b, float* __restrict__ imp) {
  int win = blockIdx.x, c = threadIdx.x;
  float s = 0.f;
  for (int n = 0; n < NTOK; ++n) s += xw[((size_t)win * NTOK + n) * CDIM + c];
  __shared__ float red[128];
  red[c] = (s * (1.f / NTOK)) * hg_w[c];
  __syncthreads();
  for (int o = 64; o; o >>= 1) { if (c < o) red[c] += red[c + o]; __syncthreads(); }
  if (c == 0) imp[win] = 1.f / (1.f + __expf(-(red[0] + hg_b[0])));
}
__global__ __launch_bounds__(128) void k_flag(const float* __restrict__ imp,
                                              int* __restrict__ flag) {
  int c = threadIdx.x;
  __shared__ float red[128];
  red[c] = imp[c];
  __syncthreads();
  for (int o = 64; o; o >>= 1) { if (c < o) red[c] += red[c + o]; __syncthreads(); }
  if (c == 0) *flag = (red[0] * (1.f / NWIN) < 0.1f) ? 1 : 0;
}

// ------------------------------ WMMA GEMM ----------------------------------
// out[m,n] = sum_k A[m,k] * W[n,k] + bias[n]  (A row-major f16, W row-major f16)
// Fragment: lane<16 -> K off 0 (halves 0..7 = k..k+7, 8..15 = k+16..k+23),
//           lane>=16 -> K off 8.  One wave = one 16x16 tile; block = 8 tiles.
struct GArgs {
  const _Float16* A; const _Float16* B; const float* bias;
  int N;
  const float* aux0; const float* aux1;
  float* of; _Float16* oh;
};

__device__ __forceinline__ v16h ldfrag(const _Float16* p) {
  v8h lo = *(const v8h*)p;
  v8h hi = *(const v8h*)(p + 16);
  v16h r;
#pragma unroll
  for (int i = 0; i < 8; ++i) { r[i] = lo[i]; r[i + 8] = hi[i]; }
  return r;
}

// EPI: 0=content gate, 1=store f32, 2=+aux0 residual -> f16,
//      4=exact GELU -> f16, 5=final: aux0 + aux1[n]*acc -> f32
template <int EPI, int K>
__global__ __launch_bounds__(256) void k_gemm(GArgs g) {
  int lane = threadIdx.x & 31, wave = threadIdx.x >> 5;
  int m0 = blockIdx.x * 128 + wave * 16;
  int n0 = blockIdx.y * 16;
  int col = n0 + (lane & 15);
  int koff = (lane & 16) ? 8 : 0;
  const _Float16* pa = g.A + (size_t)(m0 + (lane & 15)) * K + koff;
  const _Float16* pb = g.B + (size_t)col * K + koff;
  float bv = g.bias[col];
  v8f c;
#pragma unroll
  for (int i = 0; i < 8; ++i) c[i] = bv;
#pragma unroll
  for (int k0 = 0; k0 < K; k0 += 32) {
    v16h a = ldfrag(pa + k0);
    v16h b = ldfrag(pb + k0);
    c = __builtin_amdgcn_wmma_f32_16x16x32_f16(false, a, false, b, (short)0, c,
                                               false, false);
  }
  int mb = m0 + ((lane & 16) ? 8 : 0);
#pragma unroll
  for (int r = 0; r < 8; ++r) {
    size_t idx = (size_t)(mb + r) * g.N + col;
    float v = c[r];
    if (EPI == 0) {            // sigmoid(v)>0.1 hard gate on xw (aux0, f32)
      float xv = g.aux0[idx];
      float o = (v > GATE_LOGIT) ? xv : 0.f;
      g.of[idx] = o; g.oh[idx] = (_Float16)o;
    } else if (EPI == 1) {     // plain f32 store
      g.of[idx] = v;
    } else if (EPI == 2) {     // residual with gated x -> f16
      g.oh[idx] = (_Float16)(v + g.aux0[idx]);
    } else if (EPI == 4) {     // exact GELU -> f16
      g.oh[idx] = (_Float16)(0.5f * v * (1.f + erff(v * 0.70710678118654752f)));
    } else if (EPI == 5) {     // x1 + ls2*mlp -> f32 (d_out)
      g.of[idx] = g.aux0[idx] + g.aux1[col] * v;
    }
  }
}

// --------------------- attention pass 1: entropy ---------------------------
// block = (win, 8 heads); wave = one head; lane covers rows n and n+32.
__global__ __launch_bounds__(256) void k_attn_ent(const float* __restrict__ Y,
                                                  float* __restrict__ part) {
  __shared__ float kb[8][64];
  __shared__ float wacc[8];
  int lane = threadIdx.x & 31, wave = threadIdx.x >> 5;
  int win = blockIdx.x >> 4;
  int h = ((blockIdx.x & 15) << 3) + wave;
  const float* Yw = Y + (size_t)win * NTOK * QKV_LD;
  for (int m = lane; m < NTOK; m += 32) kb[wave][m] = Yw[(size_t)m * QKV_LD + 128 + h];
  __syncthreads();
  float kx = wred_max(fmaxf(kb[wave][lane], kb[wave][lane + 32]));
  float kn = wred_min(fminf(kb[wave][lane], kb[wave][lane + 32]));
  float acc = 0.f;
#pragma unroll
  for (int rr = 0; rr < 2; ++rr) {
    int n = lane + rr * 32;
    float s = Yw[(size_t)n * QKV_LD + h] * SCALE_A;
    float M = (s >= 0.f) ? s * kx : s * kn;
    float Z = 0.f, S = 0.f;
    for (int m = 0; m < NTOK; ++m) {
      float y = s * kb[wave][m] - M;
      float e = __expf(y);
      Z += e; S += y * e;
    }
    acc += __logf(Z) - S / Z;   // H = lnZ - E[y]
  }
  acc = wred_sum(acc);
  if (lane == 0) wacc[wave] = acc;
  __syncthreads();
  if (threadIdx.x == 0) {
    float t = 0.f;
    for (int i = 0; i < 8; ++i) t += wacc[i];
    part[blockIdx.x] = t;
  }
}

__global__ __launch_bounds__(256) void k_km(const float* __restrict__ part,
                                            int* __restrict__ km) {
  __shared__ float red[256];
  int tid = threadIdx.x;
  float s = 0.f;
  for (int i = tid; i < 2048; i += 256) s += part[i];
  red[tid] = s;
  __syncthreads();
  for (int o = 128; o; o >>= 1) { if (tid < o) red[tid] += red[tid + o]; __syncthreads(); }
  if (tid == 0) {
    float ent = red[0] / 1048576.0f;            // mean over B_*nh*N rows
    int k = 5 + (int)rintf(15.f * ent);         // KM_BASE + round((KM_MAX-KM_BASE)*ent)
    k = k < 5 ? 5 : (k > 20 ? 20 : k);
    *km = k;
  }
}

// ------------------ attention pass 2: top-k mask + AV ----------------------
// Monotone logits (hd==1): km_k-th largest prob per row == threshold on k_m
// itself (largest if q>0, smallest if q<0). Duplicate-aware k-th order
// statistic via iterative wave argmax/argmin on (sortable_val, lane-idx) keys.
__global__ __launch_bounds__(256) void k_attn_av(
    const float* __restrict__ Y, const float* __restrict__ gY,
    const int* __restrict__ kmp, _Float16* __restrict__ con) {
  __shared__ float kb[8][64], vb[8][64], gkb[8][64], gvb[8][64];
  int lane = threadIdx.x & 31, wave = threadIdx.x >> 5;
  int win = blockIdx.x >> 4;
  int h = ((blockIdx.x & 15) << 3) + wave;
  const float* Yw = Y + (size_t)win * NTOK * QKV_LD;
  const float* gYw = gY + (size_t)win * NTOK * QKV_LD;
  for (int m = lane; m < NTOK; m += 32) {
    kb[wave][m] = Yw[(size_t)m * QKV_LD + 128 + h];
    vb[wave][m] = Yw[(size_t)m * QKV_LD + 256 + h];
    gkb[wave][m] = gYw[(size_t)m * QKV_LD + 128 + h];
    gvb[wave][m] = gYw[(size_t)m * QKV_LD + 256 + h];
  }
  __syncthreads();
  int km = *kmp;
  float kv0 = kb[wave][lane], kv1 = kb[wave][lane + 32];
  float kx = wred_max(fmaxf(kv0, kv1));
  float kn = wred_min(fminf(kv0, kv1));
  float gkx = wred_max(fmaxf(gkb[wave][lane], gkb[wave][lane + 32]));
  float gkn = wred_min(fminf(gkb[wave][lane], gkb[wave][lane + 32]));

  unsigned long long key0 = ((unsigned long long)sortable_u(kv0) << 6) | (unsigned)lane;
  unsigned long long key1 = ((unsigned long long)sortable_u(kv1) << 6) | (unsigned)(lane + 32);
  bool r0 = false, r1 = false;
  unsigned thrHi = 0u;
  for (int i = 0; i < km; ++i) {           // km-th largest k value
    unsigned long long cand = 0ull;
    if (!r0) cand = key0;
    if (!r1 && key1 > cand) cand = key1;
#pragma unroll
    for (int o = 16; o; o >>= 1) { unsigned long long t = sx64(cand, o); if (t > cand) cand = t; }
    if (!r0 && cand == key0) r0 = true;
    else if (!r1 && cand == key1) r1 = true;
    thrHi = (unsigned)(cand >> 6);
  }
  r0 = false; r1 = false;
  unsigned thrLo = 0u;
  for (int i = 0; i < km; ++i) {           // km-th smallest k value
    unsigned long long cand = ~0ull;
    if (!r0) cand = key0;
    if (!r1 && key1 < cand) cand = key1;
#pragma unroll
    for (int o = 16; o; o >>= 1) { unsigned long long t = sx64(cand, o); if (t < cand) cand = t; }
    if (!r0 && cand == key0) r0 = true;
    else if (!r1 && cand == key1) r1 = true;
    thrLo = (unsigned)(cand >> 6);
  }

#pragma unroll
  for (int rr = 0; rr < 2; ++rr) {
    int n = lane + rr * 32;
    float s = Yw[(size_t)n * QKV_LD + h] * SCALE_A;
    float gs = gYw[(size_t)n * QKV_LD + h] * SCALE_A;
    float M = (s >= 0.f) ? s * kx : s * kn;
    float gM = (gs >= 0.f) ? gs * gkx : gs * gkn;
    float Z = 0.f, gZ = 0.f;
    for (int m = 0; m < NTOK; ++m) {
      Z += __expf(s * kb[wave][m] - M);
      gZ += __expf(gs * gkb[wave][m] - gM);
    }
    float num = 0.f, gnum = 0.f;
    for (int m = 0; m < NTOK; ++m) {
      float kvm = kb[wave][m];
      bool keep = (s > 0.f) ? (sortable_u(kvm) >= thrHi)
                : (s < 0.f) ? (sortable_u(kvm) <= thrLo) : true;
      if (keep) num += __expf(s * kvm - M) * vb[wave][m];
      gnum += __expf(gs * gkb[wave][m] - gM) * gvb[wave][m];
    }
    size_t ob = ((size_t)win * NTOK + n) * 256 + h;
    con[ob] = (_Float16)(num / Z);
    con[ob + 128] = (_Float16)(gnum / gZ);
  }
}

// ----------- window reverse + skip-select + ls1 residual + LN2 -------------
__global__ __launch_bounds__(256) void k_rev_ln2(
    const float* __restrict__ x, const float* __restrict__ aw,
    const float* __restrict__ xwf, const int* __restrict__ flag,
    const float* __restrict__ ls1, const float* __restrict__ w,
    const float* __restrict__ b, float* __restrict__ x1,
    _Float16* __restrict__ xmh) {
  int lane = threadIdx.x & 31, wave = threadIdx.x >> 5;
  int t = blockIdx.x * 8 + wave;
  int bidx = t >> 12, l = t & 4095;
  int hh = l >> 6, wc = l & 63;
  int win = bidx * 64 + (hh >> 3) * 8 + (wc >> 3);
  int n = (hh & 7) * 8 + (wc & 7);
  size_t wb = ((size_t)win * NTOK + n) * CDIM;
  int f = *flag;
  const float* src = f ? xwf : aw;   // skip path returns xw (LN1 output)
  float v[4]; float s = 0.f;
#pragma unroll
  for (int i = 0; i < 4; ++i) {
    int c = lane + 32 * i;
    float val = x[(size_t)t * CDIM + c] + ls1[c] * src[wb + c];
    x1[(size_t)t * CDIM + c] = val;
    v[i] = val; s += val;
  }
  s = wred_sum(s);
  float mu = s * (1.f / CDIM);
  float q = 0.f;
#pragma unroll
  for (int i = 0; i < 4; ++i) { float d = v[i] - mu; q += d * d; }
  q = wred_sum(q);
  float rstd = rsqrtf(q * (1.f / CDIM) + LN_EPS_F);
#pragma unroll
  for (int i = 0; i < 4; ++i) {
    int c = lane + 32 * i;
    xmh[(size_t)t * CDIM + c] = (_Float16)((v[i] - mu) * rstd * w[c] + b[c]);
  }
}

// ---------------------------------------------------------------------------
extern "C" void kernel_launch(void* const* d_in, const int* in_sizes, int n_in,
                              void* d_out, int out_size, void* d_ws,
                              size_t ws_size, hipStream_t stream) {
  (void)in_sizes; (void)n_in; (void)out_size; (void)ws_size;
  const float* x      = (const float*)d_in[0];
  const float* n1_w   = (const float*)d_in[3];
  const float* n1_b   = (const float*)d_in[4];
  const float* qkv_w  = (const float*)d_in[5];
  const float* qkv_b  = (const float*)d_in[6];
  const float* gqkv_w = (const float*)d_in[7];
  const float* gqkv_b = (const float*)d_in[8];
  const float* cg_w   = (const float*)d_in[9];
  const float* cg_b   = (const float*)d_in[10];
  const float* hg_w   = (const float*)d_in[11];
  const float* hg_b   = (const float*)d_in[12];
  const float* proj_w = (const float*)d_in[13];
  const float* proj_b = (const float*)d_in[14];
  const float* fus_w  = (const float*)d_in[15];
  const float* fus_b  = (const float*)d_in[16];
  const float* ls1    = (const float*)d_in[17];
  const float* n2_w   = (const float*)d_in[18];
  const float* n2_b   = (const float*)d_in[19];
  const float* fc1_w  = (const float*)d_in[20];
  const float* fc1_b  = (const float*)d_in[21];
  const float* fc2_w  = (const float*)d_in[22];
  const float* fc2_b  = (const float*)d_in[23];
  const float* ls2    = (const float*)d_in[24];
  float* out = (float*)d_out;

  char* p = (char*)d_ws;
  auto alloc = [&](size_t bytes) -> char* {
    char* r = p; p += (bytes + 255) & ~(size_t)255; return r;
  };
  float*    xw_f   = (float*)   alloc((size_t)M_TOK * CDIM * 4);
  _Float16* xw_h   = (_Float16*)alloc((size_t)M_TOK * CDIM * 2);
  float*    xg_f   = (float*)   alloc((size_t)M_TOK * CDIM * 4);
  _Float16* xg_h   = (_Float16*)alloc((size_t)M_TOK * CDIM * 2);
  float*    Y      = (float*)   alloc((size_t)M_TOK * QKV_LD * 4);
  float*    gYv    = (float*)   alloc((size_t)M_TOK * QKV_LD * 4);
  _Float16* con_h  = (_Float16*)alloc((size_t)M_TOK * 256 * 2);
  _Float16* tmp_h  = (_Float16*)alloc((size_t)M_TOK * CDIM * 2);
  float*    aw_f   = (float*)   alloc((size_t)M_TOK * CDIM * 4);
  float*    x1_f   = (float*)   alloc((size_t)M_TOK * CDIM * 4);
  _Float16* xm_h   = (_Float16*)alloc((size_t)M_TOK * CDIM * 2);
  _Float16* h1_h   = (_Float16*)alloc((size_t)M_TOK * 512 * 2);
  _Float16* qkv16  = (_Float16*)alloc(49152 * 2);
  _Float16* gqkv16 = (_Float16*)alloc(49152 * 2);
  _Float16* cg16   = (_Float16*)alloc(16384 * 2);
  _Float16* fus16  = (_Float16*)alloc(32768 * 2);
  _Float16* proj16 = (_Float16*)alloc(16384 * 2);
  _Float16* fc116  = (_Float16*)alloc(65536 * 2);
  _Float16* fc216  = (_Float16*)alloc(65536 * 2);
  float*    imp    = (float*)   alloc(NWIN * 4);
  float*    entP   = (float*)   alloc(2048 * 4);
  int*      flag   = (int*)     alloc(256);
  int*      km     = (int*)     alloc(256);

  // 0) weights -> f16
  k_f2h<<<192, 256, 0, stream>>>(qkv_w, qkv16, 49152);
  k_f2h<<<192, 256, 0, stream>>>(gqkv_w, gqkv16, 49152);
  k_f2h<<<64, 256, 0, stream>>>(cg_w, cg16, 16384);
  k_f2h<<<128, 256, 0, stream>>>(fus_w, fus16, 32768);
  k_f2h<<<64, 256, 0, stream>>>(proj_w, proj16, 16384);
  k_f2h<<<256, 256, 0, stream>>>(fc1_w, fc116, 65536);
  k_f2h<<<256, 256, 0, stream>>>(fc2_w, fc216, 65536);

  // 1) LN1 + window partition
  k_ln1_part<<<M_TOK / 8, 256, 0, stream>>>(x, n1_w, n1_b, xw_f, xw_h);

  // 2) hierarchical gate -> flag
  k_win_imp<<<NWIN, 128, 0, stream>>>(xw_f, hg_w, hg_b, imp);
  k_flag<<<1, 128, 0, stream>>>(imp, flag);

  // 3) content gate GEMM (EPI0): xg = xw * (sigmoid(xw@cg^T+b) > 0.1)
  { GArgs g{xw_h, cg16, cg_b, 128, xg_f, nullptr, xg_f, xg_h};
    g.aux0 = xw_f;
    k_gemm<0, 128><<<dim3(64, 8), 256, 0, stream>>>(g); }

  // 4) qkv / gqkv GEMMs (EPI1)
  { GArgs g{xg_h, qkv16, qkv_b, 384, nullptr, nullptr, Y, nullptr};
    k_gemm<1, 128><<<dim3(64, 24), 256, 0, stream>>>(g); }
  { GArgs g{xg_h, gqkv16, gqkv_b, 384, nullptr, nullptr, gYv, nullptr};
    k_gemm<1, 128><<<dim3(64, 24), 256, 0, stream>>>(g); }

  // 5) attention pass 1: entropy -> km_k
  k_attn_ent<<<2048, 256, 0, stream>>>(Y, entP);
  k_km<<<1, 256, 0, stream>>>(entP, km);

  // 6) attention pass 2: local (top-k masked) + global branch -> concat f16
  k_attn_av<<<2048, 256, 0, stream>>>(Y, gYv, km, con_h);

  // 7) fusion GEMM + residual with gated x (EPI2)
  { GArgs g{con_h, fus16, fus_b, 128, xg_f, nullptr, nullptr, tmp_h};
    k_gemm<2, 256><<<dim3(64, 8), 256, 0, stream>>>(g); }

  // 8) proj GEMM (EPI1)
  { GArgs g{tmp_h, proj16, proj_b, 128, nullptr, nullptr, aw_f, nullptr};
    k_gemm<1, 128><<<dim3(64, 8), 256, 0, stream>>>(g); }

  // 9) window reverse + skip select + ls1 residual + LN2
  k_rev_ln2<<<M_TOK / 8, 256, 0, stream>>>(x, aw_f, xw_f, flag, ls1, n2_w, n2_b,
                                           x1_f, xm_h);

  // 10) MLP: fc1 + exact GELU (EPI4), fc2 + ls2 residual -> d_out (EPI5)
  { GArgs g{xm_h, fc116, fc1_b, 512, nullptr, nullptr, nullptr, h1_h};
    k_gemm<4, 128><<<dim3(64, 32), 256, 0, stream>>>(g); }
  { GArgs g{h1_h, fc216, fc2_b, 128, x1_f, ls2, out, nullptr};
    k_gemm<5, 512><<<dim3(64, 8), 256, 0, stream>>>(g); }
}
